// DotProductNonLocalMeans_51402168599359
// MI455X (gfx1250) — matrix operand order repaired and checked by reading, not compiled, not run on hardware
//
#include <hip/hip_runtime.h>

#define B_ 16
#define C_ 128
#define N_ 16384          // H*W = 128*128
#define KSPLIT 16
#define K1_STRIDE 40      // 32 + 8 pad: conflict-free b128 LDS frag loads
#define PSTRIDE 136      // 128 + 8 pad: word-stride 68 == 4 mod 64
#define NBLK 128          // k_out column block

typedef __attribute__((ext_vector_type(16))) __bf16 bf16x16;
typedef __attribute__((ext_vector_type(8)))  __bf16 bf16x8;
typedef __attribute__((ext_vector_type(8)))  float  f32x8;
typedef __attribute__((ext_vector_type(4)))  unsigned uint4v;
typedef __attribute__((ext_vector_type(8)))  int      int8v;
typedef __attribute__((ext_vector_type(4)))  int      int4v;

// Low 32 bits of a generic pointer into __shared__ = LDS byte offset
// (shared aperture lives in the high bits on AMDGPU).
__device__ __forceinline__ unsigned ldsOff(const void* p) {
  return (unsigned)(uintptr_t)p;
}

// ASYNCcnt-tracked DMA: 16B per lane, global -> LDS, no VGPR round trip.
__device__ __forceinline__ void async_copy_b128(unsigned lds_byte_off, const float* gsrc) {
  asm volatile("global_load_async_to_lds_b128 %0, %1, off"
               :: "v"(lds_byte_off), "v"((unsigned long long)(uintptr_t)gsrc)
               : "memory");
}

__device__ __forceinline__ void wait_asynccnt0() {
  asm volatile("s_wait_asynccnt 0x0" ::: "memory");
}

// A fragment (16x32 bf16, MxK): lane<16 -> M=lane, K in {0..7,16..23};
// lane>=16 -> M=lane-16, K in {8..15,24..31}. Two 16B contiguous chunks.
__device__ __forceinline__ bf16x16 frag_a(const __bf16* base, int lane, int stride) {
  int m  = lane & 15;
  int kb = (lane >> 4) << 3;
  const __bf16* p = base + m * stride + kb;
  bf16x8 lo = *(const bf16x8*)(p);
  bf16x8 hi = *(const bf16x8*)(p + 16);
  bf16x16 a;
#pragma unroll
  for (int i = 0; i < 8; ++i) { a[i] = lo[i]; a[i + 8] = hi[i]; }
  return a;
}

// B fragment (32x16 bf16, KxN): lane<16 -> N=lane, K in {0..7,16..23}; lane>=16 ->
// N=lane-16, K in {8..15,24..31}. Column must be K-contiguous at `base + n*stride`.
__device__ __forceinline__ bf16x16 frag_b(const __bf16* base, int lane, int stride) {
  int n  = lane & 15;
  int kb = (lane >> 4) << 3;
  const __bf16* p = base + n * stride + kb;
  bf16x8 lo = *(const bf16x8*)(p);
  bf16x8 hi = *(const bf16x8*)(p + 16);
  bf16x16 b;
#pragma unroll
  for (int i = 0; i < 8; ++i) { b[i] = lo[i]; b[i + 8] = hi[i]; }
  return b;
}

// ---------------------------------------------------------------------------
// Pass 1: partial Gram matrices, async double-buffered staging.
// grid = (KSPLIT, B); 256 threads (8 waves). Chunk = 128x32 fp32.
// ---------------------------------------------------------------------------
__global__ __launch_bounds__(256) void k_gram_partial(const float* __restrict__ x,
                                                      float* __restrict__ part) {
  __shared__ float  stg[2][C_ * 32];        // fp32 staging, DMA-filled
  __shared__ __bf16 xs[C_ * K1_STRIDE];     // bf16 compute slab
  const int b = blockIdx.y, ks = blockIdx.x;
  const int tid = threadIdx.x, lane = tid & 31, wave = tid >> 5;
  const float* xb = x + (size_t)b * C_ * N_;
  const int kspan = N_ / KSPLIT;            // 1024
  const int k0 = ks * kspan;
  const int nchunks = kspan / 32;           // 32

  // per-thread staging coords: 4 x (4-float) groups
  int ci[4], kki[4];
#pragma unroll
  for (int i = 0; i < 4; ++i) { int q = tid + 256 * i; ci[i] = q >> 3; kki[i] = (q & 7) << 2; }

  // prologue: chunk 0 -> buffer 0
#pragma unroll
  for (int i = 0; i < 4; ++i)
    async_copy_b128(ldsOff(&stg[0][ci[i] * 32 + kki[i]]),
                    xb + (size_t)ci[i] * N_ + k0 + kki[i]);

  f32x8 acc[8] = {};
  for (int it = 0; it < nchunks; ++it) {
    const int buf = it & 1;
    wait_asynccnt0();                       // my chunk `it` DMAs done
    __syncthreads();                        // everyone's done; prev frag reads done
    if (it + 1 < nchunks) {                 // kick next chunk's DMA
#pragma unroll
      for (int i = 0; i < 4; ++i)
        async_copy_b128(ldsOff(&stg[buf ^ 1][ci[i] * 32 + kki[i]]),
                        xb + (size_t)ci[i] * N_ + k0 + (it + 1) * 32 + kki[i]);
    }
#pragma unroll
    for (int i = 0; i < 4; ++i) {           // convert my own staged elements
      float4 v = *(const float4*)&stg[buf][ci[i] * 32 + kki[i]];
      __bf16* d = &xs[ci[i] * K1_STRIDE + kki[i]];
      d[0] = (__bf16)v.x; d[1] = (__bf16)v.y; d[2] = (__bf16)v.z; d[3] = (__bf16)v.w;
    }
    __syncthreads();
    bf16x16 a = frag_a(xs + wave * 16 * K1_STRIDE, lane, K1_STRIDE);
#pragma unroll
    for (int t = 0; t < 8; ++t) {
      bf16x16 bb = frag_b(xs + t * 16 * K1_STRIDE, lane, K1_STRIDE); // X row == B col (Xᵀ)
      acc[t] = __builtin_amdgcn_wmma_f32_16x16x32_bf16(
          false, a, false, bb, (short)0, acc[t], false, false);
    }
  }
  float* pb = part + ((size_t)ks * B_ + b) * (C_ * C_);
  const int rbase = wave * 16 + ((lane >> 4) << 3);
  const int cf = lane & 15;
#pragma unroll
  for (int t = 0; t < 8; ++t)
#pragma unroll
    for (int v = 0; v < 8; ++v)
      pb[(rbase + v) * C_ + t * 16 + cf] = acc[t][v];
}

// ---------------------------------------------------------------------------
// Pass 2: deterministic fp32 split-K reduction, scale by 1/N, emit bf16 gram.
// ---------------------------------------------------------------------------
__global__ __launch_bounds__(256) void k_gram_reduce(const float* __restrict__ part,
                                                     __bf16* __restrict__ gram) {
  int idx = blockIdx.x * 256 + threadIdx.x;   // grid covers B_*C_*C_ exactly
  float s = 0.f;
#pragma unroll
  for (int ks = 0; ks < KSPLIT; ++ks) s += part[(size_t)ks * (B_ * C_ * C_) + idx];
  gram[idx] = (__bf16)(s * (1.0f / N_));
}

// ---------------------------------------------------------------------------
// Pass 3: m[b] = W * gram[b]  (gram symmetric => rows serve as B columns).
// ---------------------------------------------------------------------------
__global__ __launch_bounds__(256) void k_wgram(const float* __restrict__ w,
                                               const __bf16* __restrict__ gram,
                                               __bf16* __restrict__ m) {
  __shared__ __bf16 wl[C_ * PSTRIDE];
  __shared__ __bf16 gl[C_ * PSTRIDE];
  const int b = blockIdx.x, tid = threadIdx.x, lane = tid & 31, wave = tid >> 5;
  const __bf16* gb = gram + (size_t)b * C_ * C_;
#pragma unroll
  for (int i = 0; i < 64; ++i) {
    int q = tid + 256 * i;
    int r = q >> 7, c = q & 127;
    wl[r * PSTRIDE + c] = (__bf16)w[q];
    gl[r * PSTRIDE + c] = gb[q];
  }
  __syncthreads();
  f32x8 acc[8] = {};
#pragma unroll
  for (int kc = 0; kc < C_; kc += 32) {
    bf16x16 a = frag_a(wl + wave * 16 * PSTRIDE + kc, lane, PSTRIDE);
#pragma unroll
    for (int t = 0; t < 8; ++t) {
      bf16x16 bb = frag_b(gl + t * 16 * PSTRIDE + kc, lane, PSTRIDE);
      acc[t] = __builtin_amdgcn_wmma_f32_16x16x32_bf16(
          false, a, false, bb, (short)0, acc[t], false, false);
    }
  }
  __bf16* mb = m + (size_t)b * C_ * C_;
  const int rbase = wave * 16 + ((lane >> 4) << 3);
  const int cf = lane & 15;
#pragma unroll
  for (int t = 0; t < 8; ++t)
#pragma unroll
    for (int v = 0; v < 8; ++v)
      mb[(rbase + v) * C_ + t * 16 + cf] = (__bf16)acc[t][v];
}

// ---------------------------------------------------------------------------
// Pass 4: out[b] = m[b] * X[b].  grid = (N_/NBLK, B). TDM stages the X tile
// (128 rows x NBLK cols, fp32) straight into LDS; waves stage m meanwhile.
// ---------------------------------------------------------------------------
__global__ __launch_bounds__(256) void k_out(const float* __restrict__ x,
                                             const __bf16* __restrict__ m,
                                             float* __restrict__ out) {
  __shared__ float  stage[C_ * NBLK];       // [k][n] fp32, TDM-filled (64KB)
  __shared__ __bf16 xt[NBLK * PSTRIDE];     // [n][k] bf16 (B panel)
  __shared__ __bf16 ms[C_ * PSTRIDE];       // m rows, K(=d) contiguous (A panel)
  const int b = blockIdx.y;
  const int n0 = blockIdx.x * NBLK;
  const int tid = threadIdx.x, lane = tid & 31, wave = tid >> 5;

  if (wave == 0) {
    // Tensor DMA descriptor (D#), 2D: tile 128(y=k) x NBLK(x=n) of fp32.
    unsigned long long ga =
        (unsigned long long)(uintptr_t)(x + (size_t)b * C_ * N_ + n0);
    uint4v g0;
    g0[0] = 1u;                                   // count=1, user mode
    g0[1] = ldsOff(stage);                        // lds_addr (bytes)
    g0[2] = (unsigned)(ga & 0xFFFFFFFFu);         // global_addr[31:0]
    g0[3] = (unsigned)((ga >> 32) & 0x01FFFFFFu)  // global_addr[56:32]
            | (2u << 30);                         // type=2 ("image")
    int8v g1;
    g1[0] = (int)(2u << 16);                      // data_size=2 -> 4B elements
    g1[1] = (int)((unsigned)N_ << 16);            // tensor_dim0[15:0] @ bits 63:48
    g1[2] = (int)((unsigned)C_ << 16);            // tensor_dim1[15:0] @ bits 95:80
    g1[3] = (int)((unsigned)NBLK << 16);          // tile_dim0 @ bits 127:112
    g1[4] = C_;                                   // tile_dim1 @ bits 143:128
    g1[5] = N_;                                   // tensor_dim0_stride[31:0]
    g1[6] = 0;
    g1[7] = 0;
    int4v gz4 = {0, 0, 0, 0};                     // 2D: groups 2/3 unused
    int8v gz8 = {0, 0, 0, 0, 0, 0, 0, 0};
    __builtin_amdgcn_tensor_load_to_lds(g0, g1, gz4, gz4, gz8, 0);
  }
  // Stage A panel (m) while the TDM engine fills `stage`.
  const __bf16* mg = m + (size_t)b * C_ * C_;
#pragma unroll
  for (int i = 0; i < 64; ++i) {
    int q = tid + 256 * i;
    ms[(q >> 7) * PSTRIDE + (q & 127)] = mg[q];
  }
  if (wave == 0) __builtin_amdgcn_s_wait_tensorcnt(0);
  __syncthreads();

  // Convert + transpose fp32 [k][n] -> bf16 [n][k].
#pragma unroll
  for (int i = 0; i < 32; ++i) {
    int q = tid + 256 * i;                  // 0..8191
    int n = q & (NBLK - 1);
    int k2 = (q >> 7) << 1;
    float f0 = stage[k2 * NBLK + n];
    float f1 = stage[(k2 + 1) * NBLK + n];
    xt[n * PSTRIDE + k2]     = (__bf16)f0;
    xt[n * PSTRIDE + k2 + 1] = (__bf16)f1;
  }
  __syncthreads();

  float* ob = out + (size_t)b * C_ * N_ + n0;
  const int rbase = wave * 16 + ((lane >> 4) << 3);
  const int cf = lane & 15;
  f32x8 acc[8] = {};
#pragma unroll
  for (int kc = 0; kc < C_; kc += 32) {
    bf16x16 a = frag_a(ms + wave * 16 * PSTRIDE + kc, lane, PSTRIDE);
#pragma unroll
    for (int t = 0; t < 8; ++t) {
      bf16x16 bb = frag_b(xt + t * 16 * PSTRIDE + kc, lane, PSTRIDE);
      acc[t] = __builtin_amdgcn_wmma_f32_16x16x32_bf16(
          false, a, false, bb, (short)0, acc[t], false, false);
    }
  }
#pragma unroll
  for (int t = 0; t < 8; ++t) {
    int col = t * 16 + cf;
#pragma unroll
    for (int v = 0; v < 8; ++v)
      ob[(size_t)(rbase + v) * N_ + col] = acc[t][v];
  }
}

// ---------------------------------------------------------------------------
extern "C" void kernel_launch(void* const* d_in, const int* in_sizes, int n_in,
                              void* d_out, int out_size, void* d_ws, size_t ws_size,
                              hipStream_t stream) {
  (void)in_sizes; (void)n_in; (void)out_size; (void)ws_size;
  const float* x = (const float*)d_in[0];
  const float* w = (const float*)d_in[1];
  float* out = (float*)d_out;

  // Workspace layout: [KSPLIT][B][C][C] fp32 partials | bf16 gram | bf16 m
  float*  part = (float*)d_ws;
  __bf16* gram = (__bf16*)((char*)d_ws + (size_t)KSPLIT * B_ * C_ * C_ * sizeof(float));
  __bf16* mbuf = gram + (size_t)B_ * C_ * C_;

  k_gram_partial<<<dim3(KSPLIT, B_), 256, 0, stream>>>(x, part);
  k_gram_reduce<<<(B_ * C_ * C_) / 256, 256, 0, stream>>>(part, gram);
  k_wgram<<<B_, 256, 0, stream>>>(w, gram, mbuf);
  k_out<<<dim3(N_ / NBLK, B_), 256, 0, stream>>>(x, mbuf, out);
}
